// RadialNetwork2d_73083163509212
// MI455X (gfx1250) — matrix operand order repaired and checked by reading, not compiled, not run on hardware
//
#include <hip/hip_runtime.h>
#include <hip/hip_bf16.h>

typedef __attribute__((ext_vector_type(16))) _Float16 v16h;
typedef __attribute__((ext_vector_type(8)))  _Float16 v8h;
typedef __attribute__((ext_vector_type(8)))  float    v8f;

#define RBF_GRID   32
#define RBF_P      1024
#define RBF_A      8
// exp(-d2/(2*VAR)) = exp2(d2 * SCALE), VAR = 4  ->  SCALE = -0.125 * log2(e)
#define RBF_SCALE  (-0.18033688011112042f)
// 1 / (2*pi*VAR) = 1/(8*pi)
#define RBF_NORM   (0.039788735772973836f)

__global__ __launch_bounds__(256)
void RadialNetwork2d_wmma_kernel(const float* __restrict__ pos,
                                 const float* __restrict__ W,
                                 const float* __restrict__ bias,
                                 float* __restrict__ out,
                                 int nTiles)
{
    // W pre-swizzled into WMMA B-fragment order, f16, N padded 8->16:
    // Wt[((k*32 + lane)*16) + j] = (n<8) ? W[n][k*32 + ((lane>>4)*16 + j)] : 0
    // where n = lane&15. 16384 halfs = 32 KB.
    __shared__ alignas(32) _Float16 Wt[RBF_GRID * 32 * 16];
    // Per-wave separable factors: 16 rows x 32 centers each. 8 waves.
    __shared__ alignas(32) _Float16 FXs[8][16 * 32];
    __shared__ alignas(32) _Float16 FYs[8][16 * 32];

    const int tid  = threadIdx.x;
    const int lane = tid & 31;
    const int wave = tid >> 5;

    // ---- one-time per block: swizzle W into B-fragment layout (f16) ----
    for (int e = tid; e < RBF_GRID * 32 * 16; e += 256) {
        const int k  = e >> 9;         // chunk index (= cx)
        const int L  = (e >> 4) & 31;  // lane that will read this
        const int j  = e & 15;         // half index within lane fragment
        const int n  = L & 15;         // output column (action)
        const int cy = ((L >> 4) << 4) + j;   // K index within chunk
        float w = (n < RBF_A) ? W[n * RBF_P + k * RBF_GRID + cy] : 0.0f;
        Wt[e] = (_Float16)w;
    }
    __syncthreads();

    const int m = lane & 15;   // row within 16-row tile
    const int g = lane >> 4;   // lane-half selector
    const int wavesTotal = gridDim.x * 8;
    const int waveId     = blockIdx.x * 8 + wave;

    _Float16* fxRow = &FXs[wave][0];
    _Float16* fyRow = &FYs[wave][0];

    for (int tile = waveId; tile < nTiles; tile += wavesTotal) {
        const int rowBase = tile * 16;
        const float x = pos[(rowBase + m) * 2 + 0];
        const float y = pos[(rowBase + m) * 2 + 1];

        // 16 centers per lane (lane-halves split the 32 centers): 32 exp2/lane
        #pragma unroll
        for (int c16 = 0; c16 < 16; ++c16) {
            const float c  = (float)(g * 16 + c16);
            const float dx = x - c;
            const float dy = y - c;
            const float fx = RBF_NORM * exp2f(dx * dx * RBF_SCALE);
            const float fy = exp2f(dy * dy * RBF_SCALE);
            fxRow[m * 32 + g * 16 + c16] = (_Float16)fx;
            fyRow[m * 32 + g * 16 + c16] = (_Float16)fy;
        }
        // wave-local LDS visibility (lanes read values other lanes wrote)
        asm volatile("s_wait_dscnt 0x0" ::: "memory");

        // fy fragment in this kernel's contiguous K order (same as Wt):
        // halves 0..15 -> K = g*16 + j
        const v8h fy_lo = *(const v8h*)(fyRow + m * 32 + g * 16);
        const v8h fy_hi = *(const v8h*)(fyRow + m * 32 + g * 16 + 8);
        v16h fy;
        #pragma unroll
        for (int j = 0; j < 8; ++j) { fy[j] = fy_lo[j]; fy[8 + j] = fy_hi[j]; }

        v8f acc = {};
        #pragma unroll 4
        for (int k = 0; k < 32; ++k) {
            // A fragment synthesized on the fly: row-scalar fx[m][k] * fy frag
            const _Float16 fxh = fxRow[m * 32 + k];
            v16h fxs;
            #pragma unroll
            for (int j = 0; j < 16; ++j) fxs[j] = fxh;
            const v16h av = fy * fxs;                       // v_pk_mul_f16 x8
            const v16h bv = *(const v16h*)(Wt + (k * 32 + lane) * 16);
            acc = __builtin_amdgcn_wmma_f32_16x16x32_f16(
                false, av, false, bv, (short)0, acc, false, false);
        }

        // C/D layout: lane L, VGPR r -> M = g*8 + r, N = L&15. Only N<8 real.
        const int n = lane & 15;
        if (n < RBF_A) {
            const float bn = bias[n];
            #pragma unroll
            for (int r = 0; r < 8; ++r) {
                const int row = rowBase + g * 8 + r;
                out[row * RBF_A + n] = acc[r] + bn;
            }
        }
    }
}

extern "C" void kernel_launch(void* const* d_in, const int* in_sizes, int n_in,
                              void* d_out, int out_size, void* d_ws, size_t ws_size,
                              hipStream_t stream) {
    const float* pos  = (const float*)d_in[0];   // [B,2] f32
    const float* W    = (const float*)d_in[1];   // [8,1024] f32
    const float* bias = (const float*)d_in[2];   // [8] f32
    float* out = (float*)d_out;                  // [B,8] f32

    const int Bn     = in_sizes[0] / 2;
    const int nTiles = Bn / 16;                  // B = 131072 -> 8192 tiles
    int blocks = (nTiles + 31) / 32;             // 8 waves/block, ~4 tiles/wave
    if (blocks < 1) blocks = 1;
    if (blocks > 2048) blocks = 2048;

    RadialNetwork2d_wmma_kernel<<<blocks, 256, 0, stream>>>(pos, W, bias, out, nTiles);
}
